// TwoPathSigmaLowRankLinear_67156108640315
// MI455X (gfx1250) — compile-verified, matrix operand order checked by reading, not run on hardware
//
#include <hip/hip_runtime.h>

// Problem constants (match reference)
#define M_TOK 8192   // B*S
#define IN_F  4096
#define OUT_F 4096
#define RH    16
#define RL    48
#define RTOT  64
#define KSPLIT 8                  // K segments per M tile in GEMM1
#define KSEG  (IN_F / KSPLIT)     // 512

typedef __attribute__((ext_vector_type(16))) __bf16 v16bf;
typedef __attribute__((ext_vector_type(8)))  __bf16 bf16x8;
typedef __attribute__((ext_vector_type(8)))  float  v8f;
typedef __attribute__((ext_vector_type(4)))  float  f32x4;

// ---------------------------------------------------------------------------
// Kernel 0: dequantize factors to bf16 in workspace (sigma folded into V).
// ---------------------------------------------------------------------------
__global__ __launch_bounds__(256) void dequant_kernel(
    const int* __restrict__ uh_q, const float* __restrict__ uh_s,
    const int* __restrict__ vh_q, const float* __restrict__ vh_s,
    const float* __restrict__ sh,
    const int* __restrict__ ul_q, const float* __restrict__ ul_s,
    const int* __restrict__ vl_q, const float* __restrict__ vl_s,
    const float* __restrict__ sl,
    __bf16* __restrict__ Vbf, __bf16* __restrict__ Ubf) {
  int idx = blockIdx.x * blockDim.x + threadIdx.x;
  const int NV = RTOT * IN_F;
  if (idx < NV) {
    int r = idx / IN_F;
    int k = idx - r * IN_F;
    float v;
    if (r < RH) {
      v = (float)vh_q[r * IN_F + k] * vh_s[r] * sh[r];
    } else {
      int rl = r - RH;
      v = (float)vl_q[rl * IN_F + k] * vl_s[rl] * sl[rl];
    }
    Vbf[idx] = (__bf16)v;
  } else {
    int j = idx - NV;
    if (j < OUT_F * RTOT) {
      int o = j / RTOT;
      int r = j - o * RTOT;
      float v;
      if (r < RH) {
        v = (float)uh_q[o * RH + r] * uh_s[o];
      } else {
        v = (float)ul_q[o * RL + (r - RH)] * ul_s[o];
      }
      Ubf[j] = (__bf16)v;
    }
  }
}

// Load 16 bf16 (two 16B chunks) from a row into the WMMA operand layout.
__device__ __forceinline__ v16bf load_bf16_op(const __bf16* base, int c0, int c1) {
  bf16x8 lo = *(const bf16x8*)(base + c0);
  bf16x8 hi = *(const bf16x8*)(base + c1);
  v16bf r;
#pragma unroll
  for (int j = 0; j < 8; ++j) { r[j] = lo[j]; r[8 + j] = hi[j]; }
  return r;
}

// Load 16 f32 (two 32B chunks) and convert to bf16 operand layout.
__device__ __forceinline__ v16bf load_f32_op(const float* base, int c0, int c1) {
  f32x4 a0 = *(const f32x4*)(base + c0);
  f32x4 a1 = *(const f32x4*)(base + c0 + 4);
  f32x4 b0 = *(const f32x4*)(base + c1);
  f32x4 b1 = *(const f32x4*)(base + c1 + 4);
  v16bf r;
#pragma unroll
  for (int j = 0; j < 4; ++j) {
    r[j]      = (__bf16)a0[j];
    r[4 + j]  = (__bf16)a1[j];
    r[8 + j]  = (__bf16)b0[j];
    r[12 + j] = (__bf16)b1[j];
  }
  return r;
}

// ---------------------------------------------------------------------------
// Kernel 1: z[m][r] = sum_k x[m][k] * Vbf[r][k]   (M=8192, K=4096, N=R=64)
// One block (8 waves) per 16-row M tile. Each wave accumulates a 512-wide
// K segment (8-way K-split => 4096 waves device-wide), partials reduced in
// LDS, then waves 0-3 fold + convert + store one N-tile each.
// B operands are explicitly double-buffered in registers so the V loads for
// k-step i+1 issue before the wmma chain of k-step i (decouples
// global_load_b128 from its s_wait by a full iteration).
// Operand striping per CDNA5 ISA 16-bit A layout:
//   lanes 0-15 : row = lane,     K in {k0..k0+7, k0+16..k0+23}
//   lanes 16-31: row = lane-16,  K in {k0+8..k0+15, k0+24..k0+31}
// ---------------------------------------------------------------------------
__global__ __launch_bounds__(256) void gemm1_kernel(
    const float* __restrict__ x, const __bf16* __restrict__ Vbf,
    __bf16* __restrict__ zbf) {
  __shared__ float red[KSPLIT * 32 * 32];   // [wave][lane][4 accs x 8] = 32 KB

  const int wave = threadIdx.x >> 5;
  const int lane = threadIdx.x & 31;
  const int lh   = lane & 15;
  const int hi   = lane >> 4;
  const int m0   = blockIdx.x * 16;
  const int kbeg = wave * KSEG;
  const int kend = kbeg + KSEG;

  const float*  xrow = x + (size_t)(m0 + lh) * IN_F;
  const __bf16* v0   = Vbf + (size_t)(0  + lh) * IN_F;
  const __bf16* v1   = Vbf + (size_t)(16 + lh) * IN_F;
  const __bf16* v2   = Vbf + (size_t)(32 + lh) * IN_F;
  const __bf16* v3   = Vbf + (size_t)(48 + lh) * IN_F;

  v8f acc0 = {}, acc1 = {}, acc2 = {}, acc3 = {};

  // Prologue: load B tiles for the first k-step.
  int c0 = kbeg + hi * 8;
  int c1 = kbeg + 16 + hi * 8;
  v16bf b0 = load_bf16_op(v0, c0, c1);
  v16bf b1 = load_bf16_op(v1, c0, c1);
  v16bf b2 = load_bf16_op(v2, c0, c1);
  v16bf b3 = load_bf16_op(v3, c0, c1);

  for (int k0 = kbeg; k0 < kend; k0 += 32) {
    __builtin_prefetch(xrow + k0 + 256, 0, 3);  // pull x stream toward the WGP
    c0 = k0 + hi * 8;
    c1 = k0 + 16 + hi * 8;
    v16bf a = load_f32_op(xrow, c0, c1);

    // Issue next k-step's B loads before consuming the current ones
    // (last iteration wraps to kbeg; redundant in-bounds loads).
    const int kn  = (k0 + 32 < kend) ? (k0 + 32) : kbeg;
    const int n0o = kn + hi * 8;
    const int n1o = kn + 16 + hi * 8;
    v16bf nb0 = load_bf16_op(v0, n0o, n1o);
    v16bf nb1 = load_bf16_op(v1, n0o, n1o);
    v16bf nb2 = load_bf16_op(v2, n0o, n1o);
    v16bf nb3 = load_bf16_op(v3, n0o, n1o);

    acc0 = __builtin_amdgcn_wmma_f32_16x16x32_bf16(false, a, false, b0, (short)0, acc0, false, false);
    acc1 = __builtin_amdgcn_wmma_f32_16x16x32_bf16(false, a, false, b1, (short)0, acc1, false, false);
    acc2 = __builtin_amdgcn_wmma_f32_16x16x32_bf16(false, a, false, b2, (short)0, acc2, false, false);
    acc3 = __builtin_amdgcn_wmma_f32_16x16x32_bf16(false, a, false, b3, (short)0, acc3, false, false);

    b0 = nb0; b1 = nb1; b2 = nb2; b3 = nb3;
  }

  // Dump partials: 32 contiguous f32 per lane.
  float* myred = red + ((size_t)wave * 32 + lane) * 32;
#pragma unroll
  for (int j = 0; j < 8; ++j) {
    myred[0  + j] = acc0[j];
    myred[8  + j] = acc1[j];
    myred[16 + j] = acc2[j];
    myred[24 + j] = acc3[j];
  }
  __syncthreads();

  // Waves 0-3: fold the 8 partials of N-tile n == wave, convert, store z.
  if (wave < 4) {
    const int n = wave;
#pragma unroll
    for (int j = 0; j < 8; ++j) {
      float s = 0.0f;
#pragma unroll
      for (int w = 0; w < KSPLIT; ++w) {
        s += red[((size_t)w * 32 + lane) * 32 + n * 8 + j];
      }
      // C/D layout: lanes 0-15: VGPR j -> M=j, N=lane; lanes 16-31: M=8+j
      zbf[(size_t)(m0 + hi * 8 + j) * RTOT + n * 16 + lh] = (__bf16)s;
    }
  }
}

// ---------------------------------------------------------------------------
// Kernel 2: out[m][o] = sum_r z[m][r] * Ubf[o][r] + bias[o]
// K = 64 -> two WMMA k-steps. A loaded once per wave, reused over 8 N tiles.
// ---------------------------------------------------------------------------
__global__ __launch_bounds__(256) void gemm2_kernel(
    const __bf16* __restrict__ zbf, const __bf16* __restrict__ Ubf,
    const float* __restrict__ bias, float* __restrict__ out) {
  const int wave = threadIdx.x >> 5;
  const int lane = threadIdx.x & 31;
  const int lh   = lane & 15;
  const int hi   = lane >> 4;
  const int m0   = (blockIdx.x * 8 + wave) * 16;

  const __bf16* zrow = zbf + (size_t)(m0 + lh) * RTOT;
  v16bf a0 = load_bf16_op(zrow, 0  + hi * 8, 16 + hi * 8);   // r = 0..31
  v16bf a1 = load_bf16_op(zrow, 32 + hi * 8, 48 + hi * 8);   // r = 32..63

#pragma unroll
  for (int t = 0; t < 8; ++t) {
    const int n0 = (blockIdx.y * 8 + t) * 16;
    const __bf16* urow = Ubf + (size_t)(n0 + lh) * RTOT;
    v16bf b0 = load_bf16_op(urow, 0  + hi * 8, 16 + hi * 8);
    v16bf b1 = load_bf16_op(urow, 32 + hi * 8, 48 + hi * 8);
    v8f c = {};
    c = __builtin_amdgcn_wmma_f32_16x16x32_bf16(false, a0, false, b0, (short)0, c, false, false);
    c = __builtin_amdgcn_wmma_f32_16x16x32_bf16(false, a1, false, b1, (short)0, c, false, false);
    const float bv = bias[n0 + lh];
#pragma unroll
    for (int j = 0; j < 8; ++j) {
      out[(size_t)(m0 + hi * 8 + j) * OUT_F + n0 + lh] = c[j] + bv;
    }
  }
}

extern "C" void kernel_launch(void* const* d_in, const int* in_sizes, int n_in,
                              void* d_out, int out_size, void* d_ws, size_t ws_size,
                              hipStream_t stream) {
  const float* x    = (const float*)d_in[0];
  const int*   uh_q = (const int*)d_in[1];
  const float* uh_s = (const float*)d_in[2];
  const int*   vh_q = (const int*)d_in[3];
  const float* vh_s = (const float*)d_in[4];
  const float* sh   = (const float*)d_in[5];
  const int*   ul_q = (const int*)d_in[6];
  const float* ul_s = (const float*)d_in[7];
  const int*   vl_q = (const int*)d_in[8];
  const float* vl_s = (const float*)d_in[9];
  const float* sl   = (const float*)d_in[10];
  const float* bias = (const float*)d_in[11];
  float* out = (float*)d_out;

  // Workspace layout (all bf16): V' [64][4096], U [4096][64], z [8192][64]
  __bf16* Vbf = (__bf16*)d_ws;
  __bf16* Ubf = Vbf + (size_t)RTOT * IN_F;
  __bf16* zbf = Ubf + (size_t)OUT_F * RTOT;

  // 0) dequant: 64*4096 + 4096*64 = 524288 elements
  {
    int total = RTOT * IN_F + OUT_F * RTOT;
    dequant_kernel<<<(total + 255) / 256, 256, 0, stream>>>(
        uh_q, uh_s, vh_q, vh_s, sh, ul_q, ul_s, vl_q, vl_s, sl, Vbf, Ubf);
  }
  // 1) z = x * V'^T : one block per M tile, 8-way K split inside the block
  gemm1_kernel<<<M_TOK / 16, 256, 0, stream>>>(x, Vbf, zbf);
  // 2) out = z * U^T + bias : grid (64 M-groups, 32 N-groups)
  gemm2_kernel<<<dim3(M_TOK / 16 / 8, OUT_F / 16 / 8), 256, 0, stream>>>(
      zbf, Ubf, bias, out);
}
